// ScaledDotProductAttention_Operator_70394513981643
// MI455X (gfx1250) — matrix operand order repaired and checked by reading, not compiled, not run on hardware
//
#include <hip/hip_runtime.h>
#include <hip/hip_bf16.h>

typedef _Float16 v16h __attribute__((ext_vector_type(16)));
typedef float    v8f  __attribute__((ext_vector_type(8)));

constexpr int  BB = 8;
constexpr int  TT = 1024;
constexpr int  DD = 4096;            // H*W*C = 64*64*1
constexpr long OUT_ELEMS = (long)BB * TT * DD;   // first output (B,T,H,W,1)
constexpr long KX_ELEMS  = (long)BB * TT * TT;   // second output (B,T,T)

// ------------------------------------------- CDNA5 async global->LDS copy
// GLOBAL_LOAD_ASYNC_TO_LDS_B128 (GV mode): per-lane 16B global -> 16B LDS.
// Low 32 bits of a generic LDS address are the LDS byte offset (ISA 10.2).
__device__ inline void async_copy16(void* lds_dst, const void* gsrc) {
    asm volatile("global_load_async_to_lds_b128 %0, %1, off"
                 :: "v"((unsigned)(unsigned long long)lds_dst), "v"(gsrc)
                 : "memory");
}
template <int N>
__device__ inline void wait_async() {
#if __has_builtin(__builtin_amdgcn_s_wait_asynccnt)
    __builtin_amdgcn_s_wait_asynccnt(N);
#else
    asm volatile("s_wait_asynccnt %0" :: "n"(N) : "memory");
#endif
}

// ---------------------------------------------------------------- row norms
__global__ __launch_bounds__(256) void k_rownorm(const float* __restrict__ x,
                                                 float* __restrict__ out) {
    const int row = blockIdx.x;
    const float* p = x + (size_t)row * DD;
    float s = 0.f;
    for (int i = threadIdx.x; i < DD / 4; i += 256) {
        float4 v = ((const float4*)p)[i];
        s += v.x * v.x + v.y * v.y + v.z * v.z + v.w * v.w;
    }
    for (int off = 16; off; off >>= 1) s += __shfl_xor(s, off, 32);
    __shared__ float red[8];
    if ((threadIdx.x & 31) == 0) red[threadIdx.x >> 5] = s;
    __syncthreads();
    if (threadIdx.x < 8) {
        float v = red[threadIdx.x];
        v += __shfl_xor(v, 4, 32);
        v += __shfl_xor(v, 2, 32);
        v += __shfl_xor(v, 1, 32);
        if (threadIdx.x == 0) out[row] = v;
    }
}

// ------------------------------------------------- 5x5 circular conv -> f16
__global__ __launch_bounds__(256) void k_conv5(const float* __restrict__ value,
                                               const float* __restrict__ kern,
                                               _Float16* __restrict__ vc) {
    __shared__ float img[64 * 64];
    __shared__ float kw[25];
    const int bt = blockIdx.x;                    // 0 .. B*T-1
    const float* src = value + (size_t)bt * DD;
    if (threadIdx.x < 25) kw[threadIdx.x] = kern[threadIdx.x];
    for (int i = threadIdx.x; i < 4096; i += 256) img[i] = src[i];
    __syncthreads();
    for (int i = threadIdx.x; i < 4096; i += 256) {
        const int y = i >> 6, x = i & 63;
        float acc = 0.f;
#pragma unroll
        for (int dy = 0; dy < 5; ++dy) {
            const int yy = ((y + dy - 2) & 63) << 6;
#pragma unroll
            for (int dx = 0; dx < 5; ++dx) {
                const int xx = (x + dx - 2) & 63;
                acc += kw[dy * 5 + dx] * img[yy + xx];
            }
        }
        vc[(size_t)bt * DD + i] = (_Float16)acc;
    }
}

// -------------------------- f16 transpose: vc[b][t][hw] -> vc_t[b][hw][t]
__global__ __launch_bounds__(256) void k_transpose_vc(const _Float16* __restrict__ vc,
                                                      _Float16* __restrict__ vct) {
    __shared__ _Float16 tl[64][65];
    const int blk = blockIdx.x;                  // b(8) x tt(16) x ht(64)
    const int b = blk >> 10, tt = (blk >> 6) & 15, ht = blk & 63;
    const _Float16* src = vc  + ((size_t)b * TT + tt * 64) * DD + ht * 64;
    _Float16*       dst = vct + ((size_t)b * DD + ht * 64) * TT + tt * 64;
    for (int idx = threadIdx.x; idx < 64 * 64; idx += 256) {
        const int r = idx >> 6, c = idx & 63;
        tl[c][r] = src[(size_t)r * DD + c];
    }
    __syncthreads();
    for (int idx = threadIdx.x; idx < 64 * 64; idx += 256) {
        const int r = idx >> 6, c = idx & 63;
        dst[(size_t)r * TT + c] = tl[r][c];
    }
}

// ------------- fp32->f16 transpose: kx[b][i][j] -> kxT[b][j][i] (f16)
__global__ __launch_bounds__(256) void k_transpose_kx(const float* __restrict__ KX,
                                                      _Float16* __restrict__ KXT) {
    __shared__ _Float16 tl[64][65];
    const int blk = blockIdx.x;                  // b(8) x it(16) x jt(16)
    const int b = blk >> 8, it = (blk >> 4) & 15, jt = blk & 15;
    const float* src = KX  + ((size_t)b * TT + it * 64) * TT + jt * 64;
    _Float16*    dst = KXT + ((size_t)b * TT + jt * 64) * TT + it * 64;
    for (int idx = threadIdx.x; idx < 64 * 64; idx += 256) {
        const int r = idx >> 6, c = idx & 63;
        tl[c][r] = (_Float16)src[(size_t)r * TT + c];
    }
    __syncthreads();
    for (int idx = threadIdx.x; idx < 64 * 64; idx += 256) {
        const int r = idx >> 6, c = idx & 63;
        dst[(size_t)r * TT + c] = tl[r][c];
    }
}

// ----------------------------------------------- fragment loaders (wave32)
// A 16x32 f16: lane m=l&15 holds row m; half h=l>>4.
//   a[idx] : K = 16*h + idx
// B 32x16 f16: lane n=l&15 holds col n; half h=l>>4.
//   b[idx] : K = 16*h + idx
__device__ inline v16h load_fragA(const _Float16 (*S)[40], int row, int h) {
    v16h a;
#pragma unroll
    for (int g = 0; g < 8; ++g) {
        const int kk = h * 16 + 2 * g;
        a[2 * g]     = S[row][kk];
        a[2 * g + 1] = S[row][kk + 1];
    }
    return a;
}
__device__ inline v16h load_fragB(const _Float16 (*S)[40], int col, int h) {
    v16h b;
#pragma unroll
    for (int g = 0; g < 8; ++g) {
        const int kk = h * 16 + 2 * g;
        b[2 * g]     = S[col][kk];
        b[2 * g + 1] = S[col][kk + 1];
    }
    return b;
}

// ------------------------------------------ GEMM1: kx = f(Q @ K^T) epilogue
__global__ __launch_bounds__(256) void k_gemm_kx(const float* __restrict__ Q,
                                                 const float* __restrict__ Kp,
                                                 const float* __restrict__ q2,
                                                 const float* __restrict__ k2,
                                                 float* __restrict__ KX) {
    __shared__ _Float16 As[128][40];
    __shared__ _Float16 Bs[128][40];
    const int blk = blockIdx.x;
    const int b  = blk >> 6;
    const int it = (blk >> 3) & 7;
    const int jt = blk & 7;
    const float* Qb = Q  + ((size_t)b * TT + (size_t)it * 128) * DD;
    const float* Kb = Kp + ((size_t)b * TT + (size_t)jt * 128) * DD;
    const int tid = threadIdx.x, lane = tid & 31, w = tid >> 5;
    const int wm = (w & 3) * 32;       // 2 m-subtiles of 16
    const int wn = (w >> 2) * 64;      // 4 n-subtiles of 16
    const int fm = lane & 15, fh = lane >> 4;

    v8f acc[2][4] = {};
    for (int k0 = 0; k0 < DD; k0 += 32) {
        __syncthreads();
#pragma unroll
        for (int t = 0; t < 4; ++t) {
            const int idx = tid + t * 256;          // 0..1023
            const int r = idx >> 3, c = (idx & 7) * 4;
            float4 qa = *(const float4*)(Qb + (size_t)r * DD + k0 + c);
            float4 ka = *(const float4*)(Kb + (size_t)r * DD + k0 + c);
            As[r][c + 0] = (_Float16)qa.x; As[r][c + 1] = (_Float16)qa.y;
            As[r][c + 2] = (_Float16)qa.z; As[r][c + 3] = (_Float16)qa.w;
            Bs[r][c + 0] = (_Float16)ka.x; Bs[r][c + 1] = (_Float16)ka.y;
            Bs[r][c + 2] = (_Float16)ka.z; Bs[r][c + 3] = (_Float16)ka.w;
        }
        __syncthreads();
        v16h af[2], bf[4];
#pragma unroll
        for (int s = 0; s < 2; ++s) af[s] = load_fragA(As, wm + s * 16 + fm, fh);
#pragma unroll
        for (int s = 0; s < 4; ++s) bf[s] = load_fragB(Bs, wn + s * 16 + fm, fh);
#pragma unroll
        for (int i = 0; i < 2; ++i)
#pragma unroll
            for (int j = 0; j < 4; ++j)
                acc[i][j] = __builtin_amdgcn_wmma_f32_16x16x32_f16(
                    false, af[i], false, bf[j], (short)0, acc[i][j], false, false);
    }
    // epilogue: kx = rowmask * exp(-sqrt(max(q2+k2-2s,0)+eps))
    float* KXb = KX + (size_t)b * TT * TT;
    const float* q2b = q2 + b * TT;
    const float* k2b = k2 + b * TT;
    const int i_base = it * 128 + wm, j_base = jt * 128 + wn;
#pragma unroll
    for (int i = 0; i < 2; ++i)
#pragma unroll
        for (int j = 0; j < 4; ++j) {
            const int col = j_base + j * 16 + fm;
            const float k2v = k2b[col];
#pragma unroll
            for (int r = 0; r < 8; ++r) {
                const int row = i_base + i * 16 + fh * 8 + r;
                const float d2 = q2b[row] + k2v - 2.f * acc[i][j][r];
                float kxv = __expf(-__builtin_sqrtf(fmaxf(d2, 0.f) + 1e-8f));
                if (row == TT - 1) kxv = 0.f;      // M @ kx: last query row zeroed
                KXb[(size_t)row * TT + col] = kxv;
            }
        }
}

// -------- GEMM2: out = kx^T @ vc, async-LDS double-buffered (all f16 tiles)
// A tile rows = local q (j), K = i : KXT[b][j][i]
// B tile rows = local hw (n), K = i : vc_t[b][hw][i]
__global__ __launch_bounds__(256) void k_gemm_out(const _Float16* __restrict__ KXT,
                                                  const _Float16* __restrict__ vct,
                                                  float* __restrict__ out) {
    __shared__ _Float16 As[2][128][40];
    __shared__ _Float16 Bs[2][128][40];
    const int blk = blockIdx.x;
    const int b  = blk >> 8;           // 8
    const int jt = (blk >> 5) & 7;     // 8 q-tiles
    const int ht = blk & 31;           // 32 hw-tiles
    const int j0 = jt * 128, hw0 = ht * 128;
    const _Float16* Ab = KXT + (size_t)b * TT * TT + (size_t)j0 * TT;
    const _Float16* Bb = vct + (size_t)b * DD * TT + (size_t)hw0 * TT;
    const int tid = threadIdx.x, lane = tid & 31, w = tid >> 5;
    const int wm = (w & 3) * 32;
    const int wn = (w >> 2) * 64;
    const int fm = lane & 15, fh = lane >> 4;

    // staging: thread -> (row pair, 16B chunk); 4 async B128 per thread/tile
    const int sr = tid >> 2;           // 0..63
    const int sc = (tid & 3) * 8;      // halves (16B chunks)

    auto stage = [&](int buf, int i0) {
        async_copy16(&As[buf][sr][sc],      Ab + (size_t)sr * TT + i0 + sc);
        async_copy16(&As[buf][sr + 64][sc], Ab + (size_t)(sr + 64) * TT + i0 + sc);
        async_copy16(&Bs[buf][sr][sc],      Bb + (size_t)sr * TT + i0 + sc);
        async_copy16(&Bs[buf][sr + 64][sc], Bb + (size_t)(sr + 64) * TT + i0 + sc);
    };

    v8f acc[2][4] = {};
    stage(0, 0);
    for (int k = 0; k < TT / 32; ++k) {
        const int cur = k & 1;
        const bool more = (k + 1 < TT / 32);
        if (more) stage(cur ^ 1, (k + 1) * 32);   // prefetch next tile
        if (more) wait_async<4>(); else wait_async<0>();  // current tile landed
        __syncthreads();
        v16h af[2], bf[4];
#pragma unroll
        for (int s = 0; s < 2; ++s) af[s] = load_fragA(As[cur], wm + s * 16 + fm, fh);
#pragma unroll
        for (int s = 0; s < 4; ++s) bf[s] = load_fragB(Bs[cur], wn + s * 16 + fm, fh);
#pragma unroll
        for (int i = 0; i < 2; ++i)
#pragma unroll
            for (int j = 0; j < 4; ++j)
                acc[i][j] = __builtin_amdgcn_wmma_f32_16x16x32_f16(
                    false, af[i], false, bf[j], (short)0, acc[i][j], false, false);
        __syncthreads();   // protect buffer reuse vs next prefetch
    }
    float* ob = out + (size_t)b * TT * DD;
    const int q_base = j0 + wm, hw_base = hw0 + wn;
#pragma unroll
    for (int i = 0; i < 2; ++i)
#pragma unroll
        for (int j = 0; j < 4; ++j) {
            const int col = hw_base + j * 16 + fm;
#pragma unroll
            for (int r = 0; r < 8; ++r) {
                const int row = q_base + i * 16 + fh * 8 + r;
                ob[(size_t)row * DD + col] = acc[i][j][r];
            }
        }
}

// ------------------------------------------------------------------ launch
extern "C" void kernel_launch(void* const* d_in, const int* in_sizes, int n_in,
                              void* d_out, int out_size, void* d_ws, size_t ws_size,
                              hipStream_t stream) {
    const float* query = (const float*)d_in[0];
    const float* key   = (const float*)d_in[1];
    const float* value = (const float*)d_in[2];
    const float* kern  = (const float*)d_in[3];   // (1,1,5,5) = 25 floats

    float* out = (float*)d_out;                   // (B,T,H,W,1) fp32
    float* kx  = out + OUT_ELEMS;                 // (B,T,T)     fp32

    // workspace: vc f16 | vc_t f16 | kxT f16 | q2 | k2   (~151 MB peak)
    char* ws = (char*)d_ws;
    _Float16* vc  = (_Float16*)ws;
    _Float16* vct = (_Float16*)(ws + OUT_ELEMS * sizeof(_Float16));
    _Float16* kxT = (_Float16*)(ws + 2 * OUT_ELEMS * sizeof(_Float16));
    float* q2 = (float*)(ws + 2 * OUT_ELEMS * sizeof(_Float16) + KX_ELEMS * sizeof(_Float16));
    float* k2 = q2 + BB * TT;

    k_rownorm<<<BB * TT, 256, 0, stream>>>(query, q2);
    k_rownorm<<<BB * TT, 256, 0, stream>>>(key,   k2);
    k_conv5  <<<BB * TT, 256, 0, stream>>>(value, kern, vc);
    k_transpose_vc<<<BB * 16 * 64, 256, 0, stream>>>(vc, vct);
    k_gemm_kx<<<BB * 8 * 8, 256, 0, stream>>>(query, key, q2, k2, kx);
    k_transpose_kx<<<BB * 16 * 16, 256, 0, stream>>>(kx, kxT);
    k_gemm_out<<<BB * 8 * 32, 256, 0, stream>>>(kxT, vct, out);
}
